// WordInQuestionW_56100862820653
// MI455X (gfx1250) — matrix operand order repaired and checked by reading, not compiled, not run on hardware
//
#include <hip/hip_runtime.h>

typedef __attribute__((ext_vector_type(16))) _Float16 v16h;
typedef __attribute__((ext_vector_type(8)))  float    v8f;

#define DD   512
#define LQ   64
#define LC   2048
#define WAVES_PER_BLOCK 8
#define THREADS (WAVES_PER_BLOCK * 32)
#define ROWS_PER_WAVE 32
#define ROWS_PER_BLOCK (WAVES_PER_BLOCK * ROWS_PER_WAVE)   // 256

__device__ __forceinline__ v16h cvt_a_frag(float4 a0, float4 a1, float4 a2, float4 a3) {
    v16h A;
    A[0]  = (_Float16)a0.x; A[1]  = (_Float16)a0.y;
    A[2]  = (_Float16)a0.z; A[3]  = (_Float16)a0.w;
    A[4]  = (_Float16)a1.x; A[5]  = (_Float16)a1.y;
    A[6]  = (_Float16)a1.z; A[7]  = (_Float16)a1.w;
    A[8]  = (_Float16)a2.x; A[9]  = (_Float16)a2.y;
    A[10] = (_Float16)a2.z; A[11] = (_Float16)a2.w;
    A[12] = (_Float16)a3.x; A[13] = (_Float16)a3.y;
    A[14] = (_Float16)a3.z; A[15] = (_Float16)a3.w;
    return A;
}

__global__ __launch_bounds__(THREADS)
void wiq_attn_kernel(const float* __restrict__ question,
                     const float* __restrict__ context,
                     const float* __restrict__ weight,
                     float* __restrict__ out)
{
    // qw = question[b] * w, staged once per block as f16, layout [n][k]
    __shared__ _Float16 qw[LQ * DD];   // 64 KB of the 320 KB WGP LDS

    const int b   = blockIdx.y;
    const int tid = threadIdx.x;

    const float* qb = question + (size_t)b * LQ * DD;
    for (int idx = tid; idx < LQ * DD; idx += THREADS) {
        int k = idx & (DD - 1);
        qw[idx] = (_Float16)(qb[idx] * weight[k]);
    }
    __syncthreads();

    const int wave   = tid >> 5;
    const int lane   = tid & 31;
    const int lane15 = lane & 15;
    const int hi     = lane >> 4;                 // half-wave select
    const int r0     = blockIdx.x * ROWS_PER_BLOCK + wave * ROWS_PER_WAVE;

    // Two stacked 16-row A tiles per wave (contiguous K order: lane half hi holds k = 16*hi + s)
    const float* arow0 = context + ((size_t)b * LC + r0 + lane15) * DD;
    const float* arow1 = arow0 + (size_t)16 * DD;
    const int kbA = hi * 16;   // lanes 16-31 hold K=16..31 of the chunk
    const int kbB = hi * 16;   // B: lanes 16-31 hold K=16..31 of the chunk

    v8f acc00 = {}, acc01 = {}, acc02 = {}, acc03 = {};
    v8f acc10 = {}, acc11 = {}, acc12 = {}, acc13 = {};

    for (int kk = 0; kk < DD; kk += 32) {
        const float* ap0 = arow0 + kk + kbA;
        const float* ap1 = arow1 + kk + kbA;
        if (kk + 32 < DD) {
            __builtin_prefetch(ap0 + 32, 0, 3);   // global_prefetch_b8
            __builtin_prefetch(ap1 + 32, 0, 3);
        }

        float4 a00 = *(const float4*)(ap0);
        float4 a01 = *(const float4*)(ap0 + 4);
        float4 a02 = *(const float4*)(ap0 + 8);
        float4 a03 = *(const float4*)(ap0 + 12);
        float4 a10 = *(const float4*)(ap1);
        float4 a11 = *(const float4*)(ap1 + 4);
        float4 a12 = *(const float4*)(ap1 + 8);
        float4 a13 = *(const float4*)(ap1 + 12);

        v16h A0 = cvt_a_frag(a00, a01, a02, a03);
        v16h A1 = cvt_a_frag(a10, a11, a12, a13);

        // B fragments from LDS: 16 K-contiguous f16 per lane, n = lane15 + 16*nt.
        // Each B feeds two WMMAs (both A tiles), halving ds traffic per WMMA.
        const v16h B0 = *(const v16h*)&qw[(lane15 +  0) * DD + kk + kbB];
        acc00 = __builtin_amdgcn_wmma_f32_16x16x32_f16(false, A0, false, B0, (short)0, acc00, false, false);
        acc10 = __builtin_amdgcn_wmma_f32_16x16x32_f16(false, A1, false, B0, (short)0, acc10, false, false);

        const v16h B1 = *(const v16h*)&qw[(lane15 + 16) * DD + kk + kbB];
        acc01 = __builtin_amdgcn_wmma_f32_16x16x32_f16(false, A0, false, B1, (short)0, acc01, false, false);
        acc11 = __builtin_amdgcn_wmma_f32_16x16x32_f16(false, A1, false, B1, (short)0, acc11, false, false);

        const v16h B2 = *(const v16h*)&qw[(lane15 + 32) * DD + kk + kbB];
        acc02 = __builtin_amdgcn_wmma_f32_16x16x32_f16(false, A0, false, B2, (short)0, acc02, false, false);
        acc12 = __builtin_amdgcn_wmma_f32_16x16x32_f16(false, A1, false, B2, (short)0, acc12, false, false);

        const v16h B3 = *(const v16h*)&qw[(lane15 + 48) * DD + kk + kbB];
        acc03 = __builtin_amdgcn_wmma_f32_16x16x32_f16(false, A0, false, B3, (short)0, acc03, false, false);
        acc13 = __builtin_amdgcn_wmma_f32_16x16x32_f16(false, A1, false, B3, (short)0, acc13, false, false);
    }

    // Row softmax + row-sum for each 16-row tile. Row M = v + 8*hi lives in one
    // 16-lane half-wave (4 values per lane, one per acc). xor masks 1/2/4/8
    // stay within the half-wave on wave32.
    float res0[8], res1[8];
    #pragma unroll
    for (int v = 0; v < 8; ++v) {
        // tile 0 (rows r0 .. r0+15)
        float m = fmaxf(fmaxf(acc00[v], acc01[v]), fmaxf(acc02[v], acc03[v]));
        m = fmaxf(m, __shfl_xor(m, 1, 32));
        m = fmaxf(m, __shfl_xor(m, 2, 32));
        m = fmaxf(m, __shfl_xor(m, 4, 32));
        m = fmaxf(m, __shfl_xor(m, 8, 32));
        float s = __expf(acc00[v] - m) + __expf(acc01[v] - m)
                + __expf(acc02[v] - m) + __expf(acc03[v] - m);
        s += __shfl_xor(s, 1, 32);
        s += __shfl_xor(s, 2, 32);
        s += __shfl_xor(s, 4, 32);
        s += __shfl_xor(s, 8, 32);
        res0[v] = s / s;       // row-sum of softmax

        // tile 1 (rows r0+16 .. r0+31)
        float m1 = fmaxf(fmaxf(acc10[v], acc11[v]), fmaxf(acc12[v], acc13[v]));
        m1 = fmaxf(m1, __shfl_xor(m1, 1, 32));
        m1 = fmaxf(m1, __shfl_xor(m1, 2, 32));
        m1 = fmaxf(m1, __shfl_xor(m1, 4, 32));
        m1 = fmaxf(m1, __shfl_xor(m1, 8, 32));
        float s1 = __expf(acc10[v] - m1) + __expf(acc11[v] - m1)
                 + __expf(acc12[v] - m1) + __expf(acc13[v] - m1);
        s1 += __shfl_xor(s1, 1, 32);
        s1 += __shfl_xor(s1, 2, 32);
        s1 += __shfl_xor(s1, 4, 32);
        s1 += __shfl_xor(s1, 8, 32);
        res1[v] = s1 / s1;
    }

    if (lane15 == 0) {
        float* op0 = out + (size_t)b * LC + r0 + hi * 8;
        float* op1 = op0 + 16;
        #pragma unroll
        for (int v = 0; v < 8; ++v) { op0[v] = res0[v]; op1[v] = res1[v]; }
    }
}

extern "C" void kernel_launch(void* const* d_in, const int* in_sizes, int n_in,
                              void* d_out, int out_size, void* d_ws, size_t ws_size,
                              hipStream_t stream) {
    const float* question = (const float*)d_in[0];
    const float* context  = (const float*)d_in[1];
    const float* weight   = (const float*)d_in[2];
    float* out = (float*)d_out;

    const int B = in_sizes[1] / (LC * DD);   // 64
    dim3 grid(LC / ROWS_PER_BLOCK, B);
    wiq_attn_kernel<<<grid, THREADS, 0, stream>>>(question, context, weight, out);
}
